// DistanceTokenEncoder_317827580847
// MI455X (gfx1250) — matrix operand order repaired and checked by reading, not compiled, not run on hardware
//
#include <hip/hip_runtime.h>

// ---------------- problem constants (B=1, N=384, Z=128, G=128) ----------------
#define NTOK 384
#define ZDIM 128
#define KTOT 257          // G + 1 + Z features
#define KPAD 288          // padded K for 9 x (K=32) WMMA steps
#define STAG_STRIDE 296   // f16 row stride of per-wave staging (bank-spread + 16B align)
#define WCAT_STRIDE 296   // f16 col stride of transposed [w1|w2] in LDS
#define W3_STRIDE 136     // f16 col stride of transposed w3 in LDS
#define JBLK 96           // j-columns per workgroup

// LDS layout (bytes)
#define OFF_WCAT 0
#define SZ_WCAT (256 * WCAT_STRIDE * 2)          // 151552
#define OFF_W3 (OFF_WCAT + SZ_WCAT)              // 151552
#define SZ_W3 (32 * W3_STRIDE * 2)               // 8704
#define OFF_CVEC (OFF_W3 + SZ_W3)                // 160256
#define SZ_CVEC (256 * 4)                        // 1024
#define OFF_CSUM (OFF_CVEC + SZ_CVEC)            // 161280
#define SZ_CSUM (256 * 4)                        // 1024
#define OFF_STAG (OFF_CSUM + SZ_CSUM)            // 162304 (16B aligned)
#define SZ_STAG (8 * 16 * STAG_STRIDE * 2)       // 75776
#define SMEM_BYTES (OFF_STAG + SZ_STAG)          // 238080 (< 320KB WGP LDS)

typedef _Float16 v8h  __attribute__((ext_vector_type(8)));
typedef _Float16 v16h __attribute__((ext_vector_type(16)));
typedef float    v8f  __attribute__((ext_vector_type(8)));

// Two 16B LDS loads -> one 16-element f16 fragment register group.
__device__ __forceinline__ v16h ld_frag(const _Float16* p0, const _Float16* p1) {
  v8h a = *(const v8h*)p0;
  v8h b = *(const v8h*)p1;
  return __builtin_shufflevector(a, b, 0,1,2,3,4,5,6,7,8,9,10,11,12,13,14,15);
}

// Feature permutation: staging col p -> source weight row r
// p in [0,128): gaussian k=p (src row p); p in [128,256): rpe z=p-128 (src row p+1); p==256: d (src row 128)
__device__ __forceinline__ int src_row(int p) {
  return (p < 128) ? p : ((p < 256) ? (p + 1) : 128);
}

// ---------------- stage 0: r_repr[c][n][xyz] = sum_a t2b[n*4+c][a] * coords[a] ----------------
__global__ __launch_bounds__(128) void dte_rrepr(const float* __restrict__ t2b,
                                                 const float* __restrict__ coords,
                                                 float* __restrict__ rr) {
  const int mrow = blockIdx.x;          // 0..1535
  const int tid = threadIdx.x;
  float ax = 0.f, ay = 0.f, az = 0.f;
  for (int a = tid; a < 4 * NTOK; a += 128) {
    float w = t2b[(size_t)mrow * (4 * NTOK) + a];
    ax += w * coords[a * 3 + 0];
    ay += w * coords[a * 3 + 1];
    az += w * coords[a * 3 + 2];
  }
  __shared__ float red[3][128];
  red[0][tid] = ax; red[1][tid] = ay; red[2][tid] = az;
  __syncthreads();
  for (int s = 64; s > 0; s >>= 1) {
    if (tid < s) {
      red[0][tid] += red[0][tid + s];
      red[1][tid] += red[1][tid + s];
      red[2][tid] += red[2][tid + s];
    }
    __syncthreads();
  }
  if (tid == 0) {
    int c = mrow & 3, nn = mrow >> 2;
    float* o = rr + (c * NTOK + nn) * 3;
    o[0] = red[0][0]; o[1] = red[1][0]; o[2] = red[2][0];
  }
}

// ---------------- main fused kernel ----------------
__global__ __launch_bounds__(256) void dte_main(const float* __restrict__ rpe,
                                                const float* __restrict__ w1,
                                                const float* __restrict__ w2,
                                                const float* __restrict__ w3,
                                                const float* __restrict__ lnw,
                                                const float* __restrict__ lnb,
                                                const float* __restrict__ rr,
                                                float* __restrict__ out) {
  extern __shared__ char smem[];
  _Float16* WcatT = (_Float16*)(smem + OFF_WCAT);   // [256 cols][296] (K padded to 288)
  _Float16* w3T   = (_Float16*)(smem + OFF_W3);     // [32 cols][136]
  float*    cvec  = (float*)(smem + OFF_CVEC);      // [256] lnb @ W'  (bias term)
  float*    csum  = (float*)(smem + OFF_CSUM);      // [256] column sums of LN-folded W'
  _Float16* stagA = (_Float16*)(smem + OFF_STAG);   // 8 waves x [16 rows][296]

  const int tid = threadIdx.x;

  // ---- one-time per-block weight prep (LN-folded, permuted, transposed, zero-padded) ----
  for (int idx = tid; idx < 256 * KPAD; idx += 256) {
    int col = idx & 255;
    int p   = idx >> 8;                 // 0..287
    float v = 0.f;
    if (p < KTOT) {
      int r = src_row(p);
      float wsrc = (col < 128) ? w1[r * 128 + col] : w2[r * 128 + (col - 128)];
      v = wsrc * lnw[r];
    }
    WcatT[col * WCAT_STRIDE + p] = (_Float16)v;
  }
  for (int idx = tid; idx < 128 * 32; idx += 256) {
    int col = idx & 31;
    int k   = idx >> 5;
    w3T[col * W3_STRIDE + k] = (_Float16)w3[k * 32 + col];
  }
  {
    int col = tid;                      // 256 threads == 256 cols
    float sb = 0.f, sw = 0.f;
    for (int p = 0; p < KTOT; ++p) {
      int r = src_row(p);
      float wsrc = (col < 128) ? w1[r * 128 + col] : w2[r * 128 + (col - 128)];
      sb += lnb[r] * wsrc;
      sw += lnw[r] * wsrc;
    }
    cvec[col] = sb;
    csum[col] = sw;
  }
  __syncthreads();

  const int wave = tid >> 5;
  const int lane = tid & 31;
  const int m    = lane & 15;           // row within tile / N-col for B & D frags
  const int half = lane >> 4;
  _Float16* stag = stagA + wave * (16 * STAG_STRIDE);
  _Float16* row  = stag + m * STAG_STRIDE;
  const int i  = blockIdx.y;
  const int jb = blockIdx.x;

  // zero the K-padding columns [256,288) of this wave's staging once
  for (int idx = lane; idx < 16 * 32; idx += 32) {
    int r  = idx >> 5;
    int cc = 256 + (idx & 31);
    stag[r * STAG_STRIDE + cc] = (_Float16)0.f;
  }

  const float STEP  = 2.0f / 127.0f;
  const float COEFF = -0.5f / (STEP * STEP);

#pragma unroll 1
  for (int it = 0; it < 3; ++it) {
    const int t  = wave + it * 8;       // 24 tiles: c = t&3, j-subtile = t>>2
    const int c  = t & 3;
    const int jt = jb * JBLK + (t >> 2) * 16;
    const int j  = jt + m;

    // pair distance for this row (both halves compute it; only half 0 counts it as a feature)
    const float* pi = rr + (c * NTOK + i) * 3;
    const float* pj = rr + (c * NTOK + j) * 3;
    float dx = pi[0] - pj[0], dy = pi[1] - pj[1], dz = pi[2] - pj[2];
    float d  = sqrtf(dx * dx + dy * dy + dz * dz);

    float sum = 0.f, ss = 0.f;
    // pass A1: gaussian smearing -> staging cols [0,128) RAW; lane handles 64 cols
#pragma unroll
    for (int b = 0; b < 8; ++b) {
      v8h pk;
#pragma unroll
      for (int q = 0; q < 8; ++q) {
        float off = (float)(half * 64 + b * 8 + q) * STEP;
        float tt  = d - off;
        float v   = __expf(COEFF * tt * tt);
        sum += v; ss += v * v;
        pk[q] = (_Float16)v;
      }
      *(v8h*)(row + half * 64 + b * 8) = pk;
    }
    // pass A2: rpe -> staging cols [128,256) RAW
    const float4* rp = (const float4*)(rpe + ((size_t)(i * NTOK + j)) * ZDIM + half * 64);
#pragma unroll
    for (int b = 0; b < 8; ++b) {
      float4 u0 = rp[2 * b + 0];
      float4 u1 = rp[2 * b + 1];
      float vv[8] = {u0.x, u0.y, u0.z, u0.w, u1.x, u1.y, u1.z, u1.w};
      v8h pk;
#pragma unroll
      for (int q = 0; q < 8; ++q) {
        float v = vv[q];
        sum += v; ss += v * v;
        pk[q] = (_Float16)v;
      }
      *(v8h*)(row + 128 + half * 64 + b * 8) = pk;
    }
    if (half == 0) {                    // raw distance feature at col 256
      sum += d; ss += d * d;
      row[256] = (_Float16)d;
    }
    // LN stats over all 257 features of the row (2 lanes per row)
    sum += __shfl_xor(sum, 16);
    ss  += __shfl_xor(ss, 16);
    float mu   = sum * (1.0f / 257.0f);
    float var  = ss * (1.0f / 257.0f) - mu * mu;
    float rstd = rsqrtf(var + 1e-5f);
    float mr   = mu * rstd;

    // broadcast the stats of this lane's 8 D-fragment rows (M = 8*half + e)
    float rstd_e[8], mr_e[8];
#pragma unroll
    for (int e = 0; e < 8; ++e) {
      int src = 8 * half + e;           // stats for row r live in lane r
      rstd_e[e] = __shfl(rstd, src);
      mr_e[e]   = __shfl(mr, src);
    }

    // ---- stage 1 on RAW features: [16 x 288] @ [288 x 256], LN applied post-hoc ----
    v8f acc[16];
#pragma unroll
    for (int nt = 0; nt < 16; ++nt) {
#pragma unroll
      for (int e = 0; e < 8; ++e) acc[nt][e] = 0.f;
    }
#pragma unroll
    for (int s = 0; s < 9; ++s) {
      // A frag: lane(m,half), e<8 -> k = 32s + 8*half + e ; e>=8 -> +16
      v16h a = ld_frag(row + s * 32 + half * 8, row + s * 32 + half * 8 + 16);
#pragma unroll
      for (int nt = 0; nt < 16; ++nt) {
        // B frag: lane(n=m,half), e<8 -> K = 32s + 8*half + e ; e>=8 -> +16, of column nt*16+n
        const _Float16* bp = WcatT + (nt * 16 + m) * WCAT_STRIDE + s * 32 + half * 8;
        v16h bf = ld_frag(bp, bp + 16);
        acc[nt] = __builtin_amdgcn_wmma_f32_16x16x32_f16(
            false, a, false, bf, (short)0, acc[nt], false, false);
      }
    }

    // ---- post-hoc LN affine + SwiGLU; D-frags of g/u halves align elementwise ----
    //   xn@W = rstd*(x@W') - rstd*mu*csum + cvec
#pragma unroll
    for (int nt = 0; nt < 8; ++nt) {
      float cv_g = cvec[nt * 16 + m],       cs_g = csum[nt * 16 + m];
      float cv_u = cvec[(nt + 8) * 16 + m], cs_u = csum[(nt + 8) * 16 + m];
#pragma unroll
      for (int e = 0; e < 8; ++e) {
        float g  = rstd_e[e] * acc[nt][e]     + (cv_g - mr_e[e] * cs_g);
        float u  = rstd_e[e] * acc[nt + 8][e] + (cv_u - mr_e[e] * cs_u);
        float hv = (g / (1.0f + __expf(-g))) * u;
        // D elem (M=e+8*half, N=m) -> staging[h row][z col] for stage-2 A frags
        stag[(e + 8 * half) * STAG_STRIDE + nt * 16 + m] = (_Float16)hv;
      }
    }

    // ---- stage 2: [16 x 128] @ [128 x 32] ----
    v8f acc2[2];
#pragma unroll
    for (int e = 0; e < 8; ++e) { acc2[0][e] = 0.f; acc2[1][e] = 0.f; }
#pragma unroll
    for (int s = 0; s < 4; ++s) {
      v16h a = ld_frag(row + s * 32 + half * 8, row + s * 32 + half * 8 + 16);
#pragma unroll
      for (int nt = 0; nt < 2; ++nt) {
        const _Float16* bp = w3T + (nt * 16 + m) * W3_STRIDE + s * 32 + half * 8;
        v16h bf = ld_frag(bp, bp + 16);
        acc2[nt] = __builtin_amdgcn_wmma_f32_16x16x32_f16(
            false, a, false, bf, (short)0, acc2[nt], false, false);
      }
    }

    // ---- scatter: out[i, j, o*4 + c] = stage2 D(M=j-row, N=o) ----
#pragma unroll
    for (int nt = 0; nt < 2; ++nt) {
#pragma unroll
      for (int e = 0; e < 8; ++e) {
        int jj = jt + e + 8 * half;
        out[((size_t)i * NTOK + jj) * 128 + (nt * 16 + m) * 4 + c] = acc2[nt][e];
      }
    }
  }
}

extern "C" void kernel_launch(void* const* d_in, const int* in_sizes, int n_in,
                              void* d_out, int out_size, void* d_ws, size_t ws_size,
                              hipStream_t stream) {
  (void)in_sizes; (void)n_in; (void)out_size; (void)ws_size;
  const float* rpe    = (const float*)d_in[0];
  const float* t2b    = (const float*)d_in[1];
  const float* coords = (const float*)d_in[2];
  const float* lnw    = (const float*)d_in[3];
  const float* lnb    = (const float*)d_in[4];
  const float* w1     = (const float*)d_in[5];
  const float* w2     = (const float*)d_in[6];
  const float* w3     = (const float*)d_in[7];
  float* out = (float*)d_out;
  float* rr  = (float*)d_ws;                 // 4*384*3 floats of scratch

  dte_rrepr<<<dim3(4 * NTOK), dim3(128), 0, stream>>>(t2b, coords, rr);

  (void)hipFuncSetAttribute((const void*)dte_main,
                            hipFuncAttributeMaxDynamicSharedMemorySize, SMEM_BYTES);
  dte_main<<<dim3(4, NTOK), dim3(256), SMEM_BYTES, stream>>>(
      rpe, w1, w2, w3, lnw, lnb, rr, out);
}
